// AllGNN_21835613733511
// MI455X (gfx1250) — matrix operand order repaired and checked by reading, not compiled, not run on hardware
//
#include <hip/hip_runtime.h>
#include <hip/hip_bf16.h>

typedef __attribute__((ext_vector_type(16))) _Float16 v16h;
typedef __attribute__((ext_vector_type(8)))  float    v8f;

#define NNODES 12000
#define INCH   256
#define HID    64
#define NCLS   40
#define HEXT   80                        // 64 hid + 1 ones-col (deg) + 15 pad
#define KSPLIT 8                         // 375 k-tiles = 7*47 + 46
#define KTILES_TOTAL (NNODES / 32)       // 375
#define MTILES (NNODES / 16)             // 750

// -------------------------------------------------------------------------
// Fill h_extT rows 64..79: row 64 = 1.0 (ones column -> degree), rest = 0.
// h_extT is [HEXT][NNODES] f16 (transposed: column-of-h contiguous).
// -------------------------------------------------------------------------
__global__ void __launch_bounds__(256)
fill_ext_rows(_Float16* __restrict__ hT) {
    int i = blockIdx.x * 256 + threadIdx.x;       // 0 .. 16*12000-1
    int r = i / NNODES;
    int c = i % NNODES;
    hT[(size_t)(64 + r) * NNODES + c] = (r == 0) ? (_Float16)1.0f : (_Float16)0.0f;
}

// -------------------------------------------------------------------------
// Kernel A: h = x @ W_in + b_in  via WMMA f16, stored transposed as f16.
// One wave handles one 16-row M-tile, all 4 N-tiles, K = 256 (8 k-steps).
// -------------------------------------------------------------------------
__global__ void __launch_bounds__(64)
proj_kernel(const float* __restrict__ x, const float* __restrict__ Win,
            const float* __restrict__ bin, _Float16* __restrict__ hT) {
    const int wave  = threadIdx.x >> 5;
    const int lane  = threadIdx.x & 31;
    const int mt    = blockIdx.x * 2 + wave;      // 0..749
    const int m0    = mt * 16;
    const int lhalf = lane >> 4;                  // 0 or 1
    const int lmod  = lane & 15;

    v8f c[4] = {};
    const float* xrow = x + (size_t)(m0 + lmod) * INCH;

    for (int kt = 0; kt < INCH / 32; ++kt) {
        const int k0 = kt * 32;
        // ---- A tile (16x32 f16): lanes<16 hold K{0..15}, lanes>=16 K{16..31}
        const float* ap = xrow + k0 + lhalf * 16;
        float fa[16];
        *(float4*)(fa + 0)  = *(const float4*)(ap + 0);
        *(float4*)(fa + 4)  = *(const float4*)(ap + 4);
        *(float4*)(fa + 8)  = *(const float4*)(ap + 8);
        *(float4*)(fa + 12) = *(const float4*)(ap + 12);
        v16h a;
#pragma unroll
        for (int j = 0; j < 16; ++j) a[j] = (_Float16)fa[j];

        // ---- B tiles (32x16 f16): lane holds one N column, 16 consecutive K
        const int koff = k0 + lhalf * 16;
#pragma unroll
        for (int nt = 0; nt < 4; ++nt) {
            const int n = nt * 16 + lmod;
            v16h b;
#pragma unroll
            for (int j = 0; j < 16; ++j) b[j] = (_Float16)Win[(size_t)(koff + j) * HID + n];
            c[nt] = __builtin_amdgcn_wmma_f32_16x16x32_f16(
                        false, a, false, b, (short)0, c[nt], false, false);
        }
    }

    // ---- store transposed f16: lane's column is contiguous in h_extT
#pragma unroll
    for (int nt = 0; nt < 4; ++nt) {
        const int n = nt * 16 + lmod;
        const float bias = bin[n];
        union { _Float16 h[8]; uint4 u; } pk;
#pragma unroll
        for (int r = 0; r < 8; ++r) pk.h[r] = (_Float16)(c[nt][r] + bias);
        // C layout: lanes<16 rows m0..m0+7, lanes>=16 rows m0+8..15
        *(uint4*)(hT + (size_t)n * NNODES + m0 + lhalf * 8) = pk.u;
    }
}

// -------------------------------------------------------------------------
// Kernel B (hot loop): partial[ks] = adj[:, kslice] @ h_ext[kslice, :]
// One wave = one 16x80 output tile for one K-split slice. 5 WMMAs per
// k-step. adj streamed once from HBM (576 MB -> ~25us roofline);
// h_extT (1.9 MB) stays L2-resident. 6000 waves + unroll-4 keeps
// ~BW*latency (~14 MB) of adj loads in flight.
// -------------------------------------------------------------------------
__global__ void __launch_bounds__(64)
spmm_kernel(const float* __restrict__ adj, const _Float16* __restrict__ hT,
            float* __restrict__ P) {
    const int wave  = threadIdx.x >> 5;
    const int lane  = threadIdx.x & 31;
    const int mt    = blockIdx.x * 2 + wave;      // 0..749
    const int m0    = mt * 16;
    const int ks    = blockIdx.y;                 // 0..KSPLIT-1
    const int lhalf = lane >> 4;
    const int lmod  = lane & 15;

    // uneven split of 375 k-tiles: splits 0..6 get 47, split 7 gets 46
    const int ktbase = ks * 46 + (ks < 7 ? ks : 7);
    const int ktcnt  = 46 + (ks < 7 ? 1 : 0);

    v8f c[5] = {};
    const float* arow = adj + (size_t)(m0 + lmod) * NNODES;

#pragma unroll 4
    for (int kt = 0; kt < ktcnt; ++kt) {
        const int k0 = (ktbase + kt) * 32;
        const float* ap = arow + k0 + lhalf * 16;
        __builtin_prefetch(ap + 256, 0, 0);       // ~8 k-tiles ahead in this row

        float fa[16];
        *(float4*)(fa + 0)  = *(const float4*)(ap + 0);
        *(float4*)(fa + 4)  = *(const float4*)(ap + 4);
        *(float4*)(fa + 8)  = *(const float4*)(ap + 8);
        *(float4*)(fa + 12) = *(const float4*)(ap + 12);
        v16h a;
#pragma unroll
        for (int j = 0; j < 16; ++j) a[j] = (_Float16)fa[j];  // 0/1 exact in f16

        const int kbb = k0 + lhalf * 16;
#pragma unroll
        for (int nt = 0; nt < 5; ++nt) {
            const int n = nt * 16 + lmod;
            v16h b = *(const v16h*)(hT + (size_t)n * NNODES + kbb); // 32B contiguous/lane
            c[nt] = __builtin_amdgcn_wmma_f32_16x16x32_f16(
                        false, a, false, b, (short)0, c[nt], false, false);
        }
    }

    // deterministic fp32 partial store: P[ks][row][col]
    float* Pr = P + ((size_t)ks * NNODES + m0 + lhalf * 8) * HEXT;
#pragma unroll
    for (int nt = 0; nt < 5; ++nt) {
        const int n = nt * 16 + lmod;
#pragma unroll
        for (int r = 0; r < 8; ++r)
            Pr[(size_t)r * HEXT + n] = c[nt][r];
    }
}

// -------------------------------------------------------------------------
// Kernel C: fixed-order reduction over KSPLIT partials, scale by 1/(deg+1),
// tiny 64x40 classifier head with W_cls in LDS. Fully deterministic.
// Block = 320 threads handles 8 rows x 40 classes.
// -------------------------------------------------------------------------
__global__ void __launch_bounds__(320)
head_kernel(const float* __restrict__ P, const float* __restrict__ Wcls,
            const float* __restrict__ bcls, float* __restrict__ out) {
    __shared__ float sW[HID * NCLS];
    __shared__ float sB[NCLS];
    __shared__ float sA[8 * HEXT];
    const int t = threadIdx.x;

    for (int i = t; i < HID * NCLS; i += 320) sW[i] = Wcls[i];
    if (t < NCLS) sB[t] = bcls[t];

    const int r0 = blockIdx.x * 8;
    for (int i = t; i < 8 * HEXT; i += 320) {
        const int r = i / HEXT, col = i % HEXT;
        float s = 0.0f;
#pragma unroll
        for (int sp = 0; sp < KSPLIT; ++sp)
            s += P[((size_t)sp * NNODES + r0 + r) * HEXT + col];
        sA[i] = s;
    }
    __syncthreads();

    const int r = t / NCLS, cc = t % NCLS;
    const float deg = sA[r * HEXT + 64];              // adj @ ones column
    const float inv = 1.0f / (deg + 1.0f);
    float acc = 0.0f;
#pragma unroll
    for (int h = 0; h < HID; ++h) acc += sA[r * HEXT + h] * sW[h * NCLS + cc];
    out[(size_t)(r0 + r) * NCLS + cc] = acc * inv + sB[cc];
}

// -------------------------------------------------------------------------
extern "C" void kernel_launch(void* const* d_in, const int* in_sizes, int n_in,
                              void* d_out, int out_size, void* d_ws, size_t ws_size,
                              hipStream_t stream) {
    const float* x    = (const float*)d_in[0];
    const float* adj  = (const float*)d_in[1];
    const float* Win  = (const float*)d_in[2];
    const float* bin  = (const float*)d_in[3];
    const float* Wcls = (const float*)d_in[4];
    const float* bcls = (const float*)d_in[5];
    float* out = (float*)d_out;

    char* ws = (char*)d_ws;
    _Float16* hT = (_Float16*)ws;                               // [80][12000] f16, 1.92 MB
    float* P = (float*)(ws + (size_t)HEXT * NNODES * 2);        // [8][12000][80] f32, 30.7 MB

    fill_ext_rows<<<dim3((16 * NNODES) / 256), dim3(256), 0, stream>>>(hT);
    proj_kernel  <<<dim3(MTILES / 2), dim3(64), 0, stream>>>(x, Win, bin, hT);
    spmm_kernel  <<<dim3(MTILES / 2, KSPLIT), dim3(64), 0, stream>>>(adj, hT, P);
    head_kernel  <<<dim3(NNODES / 8), dim3(320), 0, stream>>>(P, Wcls, bcls, out);
}
